// MultiHeadSelfAttention_9775345566104
// MI455X (gfx1250) — matrix operand order repaired and checked by reading, not compiled, not run on hardware
//
#include <hip/hip_runtime.h>

#define D_MODEL 1024
#define N_HEADS 16
#define SEQ_LEN 2048
#define BATCH   2
#define DK      64
#define MTOK    (SEQ_LEN * BATCH)   // 4096 tokens

typedef __attribute__((ext_vector_type(16))) __bf16        v16bf;
typedef __attribute__((ext_vector_type(8)))  float         v8f;
typedef __attribute__((ext_vector_type(8)))  unsigned int  v8u;

// ---------- helpers ----------

__device__ __forceinline__ unsigned short f32_to_bf16(float f) {
    unsigned int u = __builtin_bit_cast(unsigned int, f);
    unsigned int r = u + 0x7FFFu + ((u >> 16) & 1u);      // round-to-nearest-even
    if ((u & 0x7F800000u) == 0x7F800000u) r = u;          // keep Inf/NaN
    return (unsigned short)(r >> 16);
}

// A-matrix (16x32, 16-bit) K offset for VGPR j: lanes 16-31 start at K=8,
// VGPRs 0-3 hold K pairs 0..7, VGPRs 4-7 hold K pairs 16..23 (ISA 7.12.2)
__device__ __forceinline__ int a_k0(int lane, int j) {
    return ((lane >> 4) & 1) * 8 + ((j < 4) ? (2 * j) : (16 + 2 * (j - 4)));
}
// B-matrix (32x16, 16-bit): lane = column, same K order as A (lanes 16-31 start at K=8)
__device__ __forceinline__ int b_k0(int lane, int j) {
    return ((lane >> 4) & 1) * 8 + ((j < 4) ? (2 * j) : (16 + 2 * (j - 4)));
}

// async global->LDS copy of 16 bytes per lane (GVS addressing, byte offsets)
__device__ __forceinline__ void async_copy_b128(unsigned lds_off, unsigned gbyte_off,
                                                const void* base) {
    asm volatile("global_load_async_to_lds_b128 %0, %1, %2"
                 :: "v"(lds_off), "v"(gbyte_off), "s"(base) : "memory");
}
__device__ __forceinline__ void wait_async0() {
    asm volatile("s_wait_asynccnt 0x0" ::: "memory");
}

// ---------- kernel 1: f32 -> bf16 convert (vectorized: float4 -> packed uint2) ----------

__global__ void cvt_bf16_kernel(const float4* __restrict__ src,
                                uint2* __restrict__ dst, int n4) {
    int i = blockIdx.x * blockDim.x + threadIdx.x;
    int stride = gridDim.x * blockDim.x;
    for (; i < n4; i += stride) {
        float4 f = src[i];
        uint2 o;
        o.x = (unsigned)f32_to_bf16(f.x) | ((unsigned)f32_to_bf16(f.y) << 16);
        o.y = (unsigned)f32_to_bf16(f.z) | ((unsigned)f32_to_bf16(f.w) << 16);
        dst[i] = o;
    }
}

// ---------- kernel 2: bf16 GEMM  Y = A(MxK) * Bw(NxK)^T + bias ----------
// MODE 0: write f32 to out[m*N+n]            (output projection -> d_out)
// MODE 1: write bf16 to [B][H][S][dk] layout (Q/K/V)

#define GK 32

template <int MODE>
__global__ __launch_bounds__(256) void gemm_bf16_kernel(
    const unsigned short* __restrict__ A,
    const unsigned short* __restrict__ Bw,
    const float* __restrict__ bias,
    void* __restrict__ out,
    int M, int N, int K)
{
    __shared__ __align__(16) unsigned short As[2][128 * GK];
    __shared__ __align__(16) unsigned short Bs[2][128 * GK];

    const int tid   = threadIdx.x;
    const int lane  = tid & 31;
    const int wid   = tid >> 5;
    const int waveM = wid >> 1;          // 0..3  (M direction, 32 rows each)
    const int waveN = wid & 1;           // 0..1  (N direction, 64 cols each)
    const int m0 = blockIdx.y * 128;
    const int n0 = blockIdx.x * 128;
    const int hl = lane >> 4;
    const int cl = lane & 15;

    // issue async loads of one 128x32 A tile + B tile into buffer `buf`
    auto stage = [&](int buf, int kt) {
#pragma unroll
        for (int u = 0; u < 2; ++u) {
            int i = tid + u * 256;                 // 512 chunks of 16B per tile
            int row = i >> 2, c = (i & 3) * 8;
            unsigned aoff = (unsigned)(size_t)&As[buf][row * GK + c];
            unsigned boff = (unsigned)(size_t)&Bs[buf][row * GK + c];
            unsigned ga = (unsigned)(((m0 + row) * K + kt + c) * 2);
            unsigned gb = (unsigned)(((n0 + row) * K + kt + c) * 2);
            async_copy_b128(aoff, ga, A);
            async_copy_b128(boff, gb, Bw);
        }
    };

    const v8f zerov = {0.f, 0.f, 0.f, 0.f, 0.f, 0.f, 0.f, 0.f};
    v8f acc[2][4];
    for (int i = 0; i < 2; ++i)
        for (int j = 0; j < 4; ++j) acc[i][j] = zerov;

    stage(0, 0);
    int buf = 0;
    for (int kt = 0; kt < K; kt += GK) {
        wait_async0();          // my async writes for this buffer are done
        __syncthreads();        // everyone's are; previous buffer fully consumed
        if (kt + GK < K) stage(buf ^ 1, kt + GK);

        v16bf afrag[2], bfrag[4];
        for (int i = 0; i < 2; ++i) {
            v8u u;
            int row = waveM * 32 + i * 16 + cl;
            for (int j = 0; j < 8; ++j)
                u[j] = *reinterpret_cast<const unsigned int*>(&As[buf][row * GK + a_k0(lane, j)]);
            afrag[i] = __builtin_bit_cast(v16bf, u);
        }
        for (int i = 0; i < 4; ++i) {
            v8u u;
            int row = waveN * 64 + i * 16 + cl;   // N index; fragment holds Bw[n][k] = B[k][n]
            for (int j = 0; j < 8; ++j)
                u[j] = *reinterpret_cast<const unsigned int*>(&Bs[buf][row * GK + b_k0(lane, j)]);
            bfrag[i] = __builtin_bit_cast(v16bf, u);
        }

        for (int i = 0; i < 2; ++i)
            for (int j = 0; j < 4; ++j)
                acc[i][j] = __builtin_amdgcn_wmma_f32_16x16x32_bf16(
                    false, afrag[i], false, bfrag[j], (short)0, acc[i][j], false, false);
        buf ^= 1;
    }

    // ---- epilogue: stage through LDS (reuse As, 16 KB) + coalesced b128 stores ----
    __syncthreads();   // all waves done reading As/Bs

    if constexpr (MODE == 1) {
        // bf16 out, [B][H][S][dk] layout; two 64-row passes
        unsigned short* tile = &As[0][0];        // 64 rows x 128 cols bf16
        const int mypass = waveM >> 1;
        for (int pass = 0; pass < 2; ++pass) {
            if (mypass == pass) {
                for (int i = 0; i < 2; ++i)
                    for (int j = 0; j < 4; ++j)
                        for (int r = 0; r < 8; ++r) {
                            int lrow = (waveM & 1) * 32 + i * 16 + r + hl * 8;
                            int lcol = waveN * 64 + j * 16 + cl;
                            tile[lrow * 128 + lcol] =
                                f32_to_bf16(acc[i][j][r] + bias[n0 + lcol]);
                        }
            }
            __syncthreads();
            for (int q = tid; q < 1024; q += 256) {   // 64 rows x 2 heads x 8 chunks
                int row = q >> 4;
                int sub = q & 15;
                int hh  = sub >> 3;
                int c4  = (sub & 7) * 8;
                int m   = m0 + pass * 64 + row;
                int s   = m / BATCH, b_ = m % BATCH;  // token t = s*B + b
                int h   = (n0 >> 6) + hh;
                size_t di = (((size_t)b_ * N_HEADS + h) * SEQ_LEN + s) * DK + c4;
                *reinterpret_cast<uint4*>(&reinterpret_cast<unsigned short*>(out)[di]) =
                    *reinterpret_cast<const uint4*>(&tile[row * 128 + hh * 64 + c4]);
            }
            __syncthreads();
        }
    } else {
        // f32 out, row-major; four 32-row passes
        float* tilef = reinterpret_cast<float*>(&As[0][0]);   // 32 rows x 128 f32
        for (int pass = 0; pass < 4; ++pass) {
            if (waveM == pass) {
                for (int i = 0; i < 2; ++i)
                    for (int j = 0; j < 4; ++j)
                        for (int r = 0; r < 8; ++r) {
                            int lrow = i * 16 + r + hl * 8;
                            int lcol = waveN * 64 + j * 16 + cl;
                            tilef[lrow * 128 + lcol] = acc[i][j][r] + bias[n0 + lcol];
                        }
            }
            __syncthreads();
            for (int q = tid; q < 1024; q += 256) {   // 32 rows x 32 chunks of 16B
                int row = q >> 5;
                int c4  = (q & 31) * 4;
                int m   = m0 + pass * 32 + row;
                *reinterpret_cast<uint4*>(
                    &reinterpret_cast<float*>(out)[(size_t)m * N + n0 + c4]) =
                    *reinterpret_cast<const uint4*>(&tilef[row * 128 + c4]);
            }
            __syncthreads();
        }
    }
}

// ---------- kernel 3: flash attention over one (b,h), 128 query rows/block ----------

__global__ __launch_bounds__(256) void flash_attn_kernel(
    const unsigned short* __restrict__ Q,   // [B*H][S][DK] bf16
    const unsigned short* __restrict__ Km,
    const unsigned short* __restrict__ Vm,
    unsigned short* __restrict__ ctx)       // [MTOK][D_MODEL] bf16 (token-major)
{
    __shared__ __align__(16) unsigned short Ks[2][64 * 64];
    __shared__ __align__(16) unsigned short Vs[2][64 * 64];
    __shared__ __align__(16) unsigned short Ps[8][16 * 64];   // per-wave P scratch

    const int tid  = threadIdx.x;
    const int lane = tid & 31;
    const int wid  = tid >> 5;
    const int hl   = lane >> 4;
    const int cl   = lane & 15;
    const int bh   = blockIdx.y;           // b*H + h
    const int q0   = blockIdx.x * 128;

    const unsigned short* Qb = Q  + (size_t)bh * SEQ_LEN * DK;
    const unsigned short* Kb = Km + (size_t)bh * SEQ_LEN * DK;
    const unsigned short* Vb = Vm + (size_t)bh * SEQ_LEN * DK;

    // issue async loads of 64x64 K and V tiles for key-tile kt into buffer `buf`
    auto stage_kv = [&](int buf, int kt) {
#pragma unroll
        for (int u = 0; u < 2; ++u) {
            int i = tid + u * 256;                 // 512 chunks of 16B per tile
            int row = i >> 3, c = (i & 7) * 8;
            unsigned koff = (unsigned)(size_t)&Ks[buf][row * 64 + c];
            unsigned voff = (unsigned)(size_t)&Vs[buf][row * 64 + c];
            unsigned g = (unsigned)((((kt * 64 + row)) * DK + c) * 2);
            async_copy_b128(koff, g, Kb);
            async_copy_b128(voff, g, Vb);
        }
    };

    // Q A-fragments for this wave's 16 rows, kept in registers all kernel
    v16bf qfrag[2];
    {
        int qrow = q0 + wid * 16 + cl;
        for (int i = 0; i < 2; ++i) {
            v8u u;
            for (int j = 0; j < 8; ++j) {
                int k0 = i * 32 + a_k0(lane, j);
                u[j] = *reinterpret_cast<const unsigned int*>(&Qb[(size_t)qrow * DK + k0]);
            }
            qfrag[i] = __builtin_bit_cast(v16bf, u);
        }
    }

    const v8f zerov = {0.f, 0.f, 0.f, 0.f, 0.f, 0.f, 0.f, 0.f};
    v8f acc[4];
    for (int n = 0; n < 4; ++n) acc[n] = zerov;
    float mrow[8], lrow[8];
    for (int r = 0; r < 8; ++r) { mrow[r] = -INFINITY; lrow[r] = 0.f; }

    const int nkt = q0 / 64 + 2;           // causal: keys up to q0+127
    stage_kv(0, 0);
    int buf = 0;
    for (int kt = 0; kt < nkt; ++kt) {
        wait_async0();
        __syncthreads();
        if (kt + 1 < nkt) stage_kv(buf ^ 1, kt + 1);

        // scores S = Q * K^T   (16 rows x 64 keys, K-dim = dk = 64)
        v8f sacc[4];
        for (int n = 0; n < 4; ++n) {
            v8f s = zerov;
            int key = n * 16 + cl;         // fragment column = key
            for (int half = 0; half < 2; ++half) {
                v8u u;
                for (int j = 0; j < 8; ++j) {
                    int kd = half * 32 + b_k0(lane, j);    // B[k=kd][n=key] = K[key][kd]
                    u[j] = *reinterpret_cast<const unsigned int*>(&Ks[buf][key * 64 + kd]);
                }
                v16bf kf = __builtin_bit_cast(v16bf, u);
                s = __builtin_amdgcn_wmma_f32_16x16x32_bf16(
                        false, qfrag[half], false, kf, (short)0, s, false, false);
            }
            sacc[n] = s;
        }

        // online softmax (row = r + 8*hl, replicated across the 16-lane N group)
        float sc[4][8];
        for (int r = 0; r < 8; ++r) {
            int qrow = q0 + wid * 16 + r + hl * 8;
            float mx = -INFINITY;
            for (int n = 0; n < 4; ++n) {
                int key = kt * 64 + n * 16 + cl;
                float v = sacc[n][r] * 0.125f;              // 1/sqrt(dk)
                if (key > qrow) v = -INFINITY;              // causal
                sc[n][r] = v;
                mx = fmaxf(mx, v);
            }
            for (int off = 1; off < 16; off <<= 1)
                mx = fmaxf(mx, __shfl_xor(mx, off, 32));
            float mnew = fmaxf(mrow[r], mx);
            float c = __expf(mrow[r] - mnew);
            float rs = 0.f;
            for (int n = 0; n < 4; ++n) {
                float p = __expf(sc[n][r] - mnew);
                sc[n][r] = p;
                rs += p;
            }
            for (int off = 1; off < 16; off <<= 1)
                rs += __shfl_xor(rs, off, 32);
            lrow[r] = lrow[r] * c + rs;
            mrow[r] = mnew;
            for (int n = 0; n < 4; ++n) acc[n][r] *= c;
        }

        // spill P (bf16) to per-wave LDS, reload in A-fragment layout
        for (int n = 0; n < 4; ++n)
            for (int r = 0; r < 8; ++r)
                Ps[wid][(r + hl * 8) * 64 + n * 16 + cl] = f32_to_bf16(sc[n][r]);

        // ctx += P (16x64) * V (64x64), K-dim = keys
        for (int half = 0; half < 2; ++half) {
            v8u ua;
            for (int j = 0; j < 8; ++j) {
                int k0 = half * 32 + a_k0(lane, j);
                ua[j] = *reinterpret_cast<const unsigned int*>(&Ps[wid][cl * 64 + k0]);
            }
            v16bf pa = __builtin_bit_cast(v16bf, ua);
            for (int n = 0; n < 4; ++n) {
                v8u ub;
                int d = n * 16 + cl;
                for (int j = 0; j < 8; ++j) {
                    int k0 = half * 32 + b_k0(lane, j);     // B[k=key][n=d] = V[key][d]
                    unsigned int lo = Vs[buf][(k0    ) * 64 + d];
                    unsigned int hi = Vs[buf][(k0 + 1) * 64 + d];
                    ub[j] = lo | (hi << 16);
                }
                v16bf vf = __builtin_bit_cast(v16bf, ub);
                acc[n] = __builtin_amdgcn_wmma_f32_16x16x32_bf16(
                            false, pa, false, vf, (short)0, acc[n], false, false);
            }
        }
        buf ^= 1;
    }

    // normalize and write ctx (token-major, column h*64 + d)
    const int b_ = bh / N_HEADS;
    const int h  = bh % N_HEADS;
    for (int n = 0; n < 4; ++n)
        for (int r = 0; r < 8; ++r) {
            int qrow = q0 + wid * 16 + r + hl * 8;
            float v = acc[n][r] / lrow[r];
            size_t t = (size_t)qrow * BATCH + b_;
            ctx[t * D_MODEL + h * 64 + n * 16 + cl] = f32_to_bf16(v);
        }
}

// ---------- host ----------

extern "C" void kernel_launch(void* const* d_in, const int* in_sizes, int n_in,
                              void* d_out, int out_size, void* d_ws, size_t ws_size,
                              hipStream_t stream) {
    (void)in_sizes; (void)n_in; (void)out_size; (void)ws_size;

    const float* x  = (const float*)d_in[0];
    const float* Wq = (const float*)d_in[1];
    const float* bq = (const float*)d_in[2];
    const float* Wk = (const float*)d_in[3];
    const float* bk = (const float*)d_in[4];
    const float* Wv = (const float*)d_in[5];
    const float* bv = (const float*)d_in[6];
    const float* Wo = (const float*)d_in[7];
    const float* bo = (const float*)d_in[8];
    float* out = (float*)d_out;

    char* ws = (char*)d_ws;
    const size_t MB = 1u << 20;
    unsigned short* xb   = (unsigned short*)(ws);              //  8 MB: x bf16, token-major
    unsigned short* wqb  = (unsigned short*)(ws +  8 * MB);    //  2 MB
    unsigned short* wkb  = (unsigned short*)(ws + 10 * MB);
    unsigned short* wvb  = (unsigned short*)(ws + 12 * MB);
    unsigned short* wob  = (unsigned short*)(ws + 14 * MB);
    unsigned short* Qb   = (unsigned short*)(ws + 16 * MB);    //  8 MB [B][H][S][dk]
    unsigned short* Kb   = (unsigned short*)(ws + 24 * MB);
    unsigned short* Vb   = (unsigned short*)(ws + 32 * MB);
    unsigned short* ctxb = (unsigned short*)(ws + 40 * MB);    //  8 MB token-major

    cvt_bf16_kernel<<<1024, 256, 0, stream>>>((const float4*)x,  (uint2*)xb,
                                              MTOK * D_MODEL / 4);
    cvt_bf16_kernel<<<512, 256, 0, stream>>>((const float4*)Wq, (uint2*)wqb,
                                             D_MODEL * D_MODEL / 4);
    cvt_bf16_kernel<<<512, 256, 0, stream>>>((const float4*)Wk, (uint2*)wkb,
                                             D_MODEL * D_MODEL / 4);
    cvt_bf16_kernel<<<512, 256, 0, stream>>>((const float4*)Wv, (uint2*)wvb,
                                             D_MODEL * D_MODEL / 4);
    cvt_bf16_kernel<<<512, 256, 0, stream>>>((const float4*)Wo, (uint2*)wob,
                                             D_MODEL * D_MODEL / 4);

    dim3 gg(D_MODEL / 128, MTOK / 128);   // (8, 32)
    gemm_bf16_kernel<1><<<gg, 256, 0, stream>>>(xb, wqb, bq, Qb, MTOK, D_MODEL, D_MODEL);
    gemm_bf16_kernel<1><<<gg, 256, 0, stream>>>(xb, wkb, bk, Kb, MTOK, D_MODEL, D_MODEL);
    gemm_bf16_kernel<1><<<gg, 256, 0, stream>>>(xb, wvb, bv, Vb, MTOK, D_MODEL, D_MODEL);

    dim3 ga(SEQ_LEN / 128, BATCH * N_HEADS);  // (16, 32)
    flash_attn_kernel<<<ga, 256, 0, stream>>>(Qb, Kb, Vb, ctxb);

    gemm_bf16_kernel<0><<<gg, 256, 0, stream>>>(ctxb, wob, bo, out, MTOK, D_MODEL, D_MODEL);
}
